// SpatialLinearAttention_1443109012364
// MI455X (gfx1250) — matrix operand order repaired and checked by reading, not compiled, not run on hardware
//
#include <hip/hip_runtime.h>
#include <hip/hip_bf16.h>

// ---------------------------------------------------------------------------
// SpatialLinearAttention for MI455X (gfx1250, wave32, WMMA bf16)
// Pipeline: qkv GEMM (+q softmax over d) -> k row stats -> context GEMM
//           -> fused (ctx^T @ q') then w_out GEMM + bias, all bf16 WMMA.
// All WMMA operands are staged in LDS in *fragment order*: each lane's 16
// bf16 elements are contiguous (32B aligned) so a fragment load is 2x
// ds_load_b128 instead of 16 strided 16-bit gathers.
// Workspace: qb/kb/vb bf16 (3 x 64MiB) + stats + ctx (~203 MB assumed).
// ---------------------------------------------------------------------------

typedef __attribute__((ext_vector_type(16))) __bf16 v16bf;
typedef __attribute__((ext_vector_type(8)))  float  v8f;

#define NT1 256   // n tile for kernel 1

__device__ __forceinline__ __bf16 to_bf(float f) { return (__bf16)f; }

__device__ __forceinline__ unsigned pack_bf2(float a, float b) {
    union { __bf16 h[2]; unsigned u; } x;
    x.h[0] = (__bf16)a; x.h[1] = (__bf16)b; return x.u;
}
__device__ __forceinline__ unsigned pack2h(__bf16 a, __bf16 b) {
    union { __bf16 h[2]; unsigned u; } x;
    x.h[0] = a; x.h[1] = b; return x.u;
}
union BF8 { uint4 u4; __bf16 h[8]; };

// Fragment-order conventions (wave32):
//  A (16x32, MxK):  lane l<16 holds M=l, elements e: k = hi*16 + e
//                   (hi = lane>=16).  Stored so frag = 16 contiguous bf16/lane.
//  B (32x16, KxN):  lane l<16 holds N=l, k = hi*16 + e.
//  C/D (16x16):     lane N = l, VGPR r -> M = hi*8 + r.

// ---------------------------------------------------------------------------
// Kernel 1: qkv = w_qkv @ x (per bf slice), q softmax over d in-block.
// grid (16 ntiles, 24 rowgroups, 32 bf), block 256 (8 waves).
// ---------------------------------------------------------------------------
__global__ __launch_bounds__(256) void k_qkv(const float* __restrict__ x,
                                             const float* __restrict__ wqkv,
                                             __bf16* __restrict__ qb,
                                             __bf16* __restrict__ kb,
                                             __bf16* __restrict__ vb) {
    __shared__ __align__(32) __bf16 Bsw[16 * 32 * 16]; // [nsub][lane][e] 16KB
    __shared__ __align__(32) __bf16 Asw[2 * 32 * 16];  // [mt][lane][e]    2KB
    __shared__ float Cs[32 * (NT1 + 4)];               // [m][n]          33KB

    const int tid   = threadIdx.x;
    const int ntile = blockIdx.x;
    const int rg    = blockIdx.y;
    const int bfi   = blockIdx.z;
    const int n0t   = ntile * NT1;
    const int R0    = rg * 32;
    const int b     = bfi >> 4, f = bfi & 15;
    const float* xb = x + (size_t)b * 16777216 + (size_t)f * 4096; // xb[c*65536+n]

    const int wave = tid >> 5, lane = tid & 31;
    const int mt   = wave & 1;          // m tile (0/1) -> m_off = mt*16
    const int wn   = wave >> 1;         // 0..3 -> 64-col chunk

    // B staging: this thread owns column n = n0t + tid for all 32 k
    const int bsub = tid >> 4, bl = tid & 15;
    __bf16* bdst0 = &Bsw[((bsub * 32) + bl) * 16];       // k 0..15
    __bf16* bdst1 = &Bsw[((bsub * 32) + 16 + bl) * 16];  // k 16..31
    const float* xcol = xb + n0t + tid;

    // A staging task: row of w_qkv + one float4 of its 32-k chunk
    const int arow = tid >> 3, aq = tid & 7;
    const int ahi = aq & 1, ahalf = (aq >> 1) & 1, ak4 = aq >> 2;
    const int akoff = ahi * 16 + ahalf * 8 + ak4 * 4;    // k within chunk
    const int aeoff = ahalf * 8 + ak4 * 4;               // fragment element
    __bf16* adst = &Asw[(((arow >> 4) * 32) + (arow & 15) + ahi * 16) * 16 + aeoff];

    v8f acc[4];
#pragma unroll
    for (int t = 0; t < 4; ++t) acc[t] = v8f{};

    for (int k0 = 0; k0 < 256; k0 += 32) {
        // ---- stage X tile column (32 coalesced b32 loads) ----
        float xv[32];
        const float* xc = xcol + (size_t)k0 * 65536;
#pragma unroll
        for (int kk = 0; kk < 32; ++kk) xv[kk] = xc[(size_t)kk * 65536];
        {
            uint4 s0, s1;
            s0.x = pack_bf2(xv[0], xv[1]);   s0.y = pack_bf2(xv[2], xv[3]);
            s0.z = pack_bf2(xv[4], xv[5]);   s0.w = pack_bf2(xv[6], xv[7]);
            s1.x = pack_bf2(xv[8], xv[9]);   s1.y = pack_bf2(xv[10], xv[11]);
            s1.z = pack_bf2(xv[12], xv[13]); s1.w = pack_bf2(xv[14], xv[15]);
            ((uint4*)bdst0)[0] = s0; ((uint4*)bdst0)[1] = s1;
            s0.x = pack_bf2(xv[16], xv[17]); s0.y = pack_bf2(xv[18], xv[19]);
            s0.z = pack_bf2(xv[20], xv[21]); s0.w = pack_bf2(xv[22], xv[23]);
            s1.x = pack_bf2(xv[24], xv[25]); s1.y = pack_bf2(xv[26], xv[27]);
            s1.z = pack_bf2(xv[28], xv[29]); s1.w = pack_bf2(xv[30], xv[31]);
            ((uint4*)bdst1)[0] = s0; ((uint4*)bdst1)[1] = s1;
        }
        // ---- stage W tile (float4 -> 8B fragment-order store) ----
        {
            const float4 v = *(const float4*)(wqkv + (size_t)(R0 + arow) * 256 + k0 + akoff);
            uint2 st; st.x = pack_bf2(v.x, v.y); st.y = pack_bf2(v.z, v.w);
            *(uint2*)adst = st;
        }
        if (k0 + 32 < 256)  // global_prefetch_b8 of next X chunk
            __builtin_prefetch(xcol + (size_t)(k0 + 32) * 65536, 0, 1);
        __syncthreads();

        const v16bf afrag = *(const v16bf*)&Asw[(mt * 32 + lane) * 16];
#pragma unroll
        for (int t = 0; t < 4; ++t) {
            const v16bf bfrag = *(const v16bf*)&Bsw[((wn * 4 + t) * 32 + lane) * 16];
            acc[t] = __builtin_amdgcn_wmma_f32_16x16x32_bf16(
                false, afrag, false, bfrag, (short)0, acc[t], false, false);
        }
        __syncthreads();
    }

    // stage accumulators into Cs [m][n]
    {
        const int l = lane & 15, hi = lane >> 4;
#pragma unroll
        for (int t = 0; t < 4; ++t)
#pragma unroll
            for (int r = 0; r < 8; ++r)
                Cs[(mt * 16 + hi * 8 + r) * (NT1 + 4) + wn * 64 + t * 16 + l] = acc[t][r];
    }
    __syncthreads();

    const size_t obase = (size_t)bfi * (256 * 4096);
    if (rg < 8) {
        const int col = tid;
        float mx = -3.0e38f;
#pragma unroll
        for (int d = 0; d < 32; ++d) mx = fmaxf(mx, Cs[d * (NT1 + 4) + col]);
        float s = 0.f;
#pragma unroll
        for (int d = 0; d < 32; ++d) s += __expf(Cs[d * (NT1 + 4) + col] - mx);
        const float inv = 0.17677669529663687f / s;  // DIM_HEAD^-0.5 / sum
#pragma unroll
        for (int d = 0; d < 32; ++d)
            qb[obase + (size_t)(R0 + d) * 4096 + n0t + col] =
                to_bf(__expf(Cs[d * (NT1 + 4) + col] - mx) * inv);
    } else if (rg < 16) {
        const int Rk = R0 - 256;
#pragma unroll
        for (int d = 0; d < 32; ++d)
            kb[obase + (size_t)(Rk + d) * 4096 + n0t + tid] = to_bf(Cs[d * (NT1 + 4) + tid]);
    } else {
        const int Rv = R0 - 512;
#pragma unroll
        for (int d = 0; d < 32; ++d)
            vb[obase + (size_t)(Rv + d) * 4096 + n0t + tid] = to_bf(Cs[d * (NT1 + 4) + tid]);
    }
}

// ---------------------------------------------------------------------------
// Kernel 2: per-row max and sum(exp) over n=4096 of k.  grid 8192, block 256
// ---------------------------------------------------------------------------
__global__ __launch_bounds__(256) void k_stats(const __bf16* __restrict__ kb,
                                               float* __restrict__ kmax,
                                               float* __restrict__ ksum) {
    __shared__ float red[256];
    const int row = blockIdx.x;
    const int tid = threadIdx.x;
    const uint4* s4 = (const uint4*)(kb + (size_t)row * 4096);

    BF8 a0, a1;
    a0.u4 = s4[tid]; a1.u4 = s4[tid + 256];
    float v[16];
    float mx = -3.0e38f;
#pragma unroll
    for (int i = 0; i < 8; ++i) {
        v[i] = (float)a0.h[i]; v[8 + i] = (float)a1.h[i];
    }
#pragma unroll
    for (int i = 0; i < 16; ++i) mx = fmaxf(mx, v[i]);
    red[tid] = mx; __syncthreads();
    for (int s = 128; s > 0; s >>= 1) {
        if (tid < s) red[tid] = fmaxf(red[tid], red[tid + s]);
        __syncthreads();
    }
    mx = red[0]; __syncthreads();
    float sm = 0.f;
#pragma unroll
    for (int i = 0; i < 16; ++i) sm += __expf(v[i] - mx);
    red[tid] = sm; __syncthreads();
    for (int s = 128; s > 0; s >>= 1) {
        if (tid < s) red[tid] += red[tid + s];
        __syncthreads();
    }
    if (tid == 0) { kmax[row] = mx; ksum[row] = red[0]; }
}

// ---------------------------------------------------------------------------
// Kernel 3: context[d][e] = sum_n softmax_n(k)[d][n] * v[e][n]  per (bf, head)
// grid 256, block 256.  n chunk = 64, split across wave groups.
// ---------------------------------------------------------------------------
__global__ __launch_bounds__(256) void k_ctx(const __bf16* __restrict__ kb,
                                             const __bf16* __restrict__ vb,
                                             const float* __restrict__ kmax,
                                             const float* __restrict__ ksum,
                                             float* __restrict__ ctx) {
    __shared__ __align__(32) __bf16 Ksw[2 * 2 * 32 * 16]; // [kc][mt][lane][e]
    __shared__ __align__(32) __bf16 Vsw[2 * 2 * 32 * 16]; // [kc][s][lane][e]
    __shared__ float Cp[2 * 32 * 36];

    const int tid = threadIdx.x;
    const int h   = blockIdx.x & 7;
    const int bfi = blockIdx.x >> 3;
    const size_t base = (size_t)bfi * (256 * 4096) + (size_t)h * 32 * 4096;
    const int srow = bfi * 256 + h * 32;

    const int wave = tid >> 5, lane = tid & 31;
    const int t  = wave & 3;
    const int mt = t & 1, s = t >> 1;
    const int kc = wave >> 2;

    // staging tasks: K-operand (A): row m of exp(k); V-operand (B): column e
    const int km = tid >> 3, koct = tid & 7;
    const int kkc = koct >> 2, khi = (koct >> 1) & 1, keb = (koct & 1) * 8;
    __bf16* kdst = &Ksw[(((kkc * 2 + (km >> 4)) * 32) + (km & 15) + khi * 16) * 16 + keb];
    const int vec = tid >> 3, voct = tid & 7;
    const int vkc = voct >> 2, vhi = (voct >> 1) & 1, veb = (voct & 1) * 8;
    __bf16* vdst = &Vsw[(((vkc * 2 + (vec >> 4)) * 32) + vhi * 16 + (vec & 15)) * 16 + veb];

    const float mxr = kmax[srow + km];

    v8f acc = v8f{};
    for (int c0 = 0; c0 < 4096; c0 += 64) {
        {   // exp(k - rowmax) -> Ksw, 8 bf16 per thread (b128 in, b128 out)
            BF8 kv; kv.u4 = *(const uint4*)(kb + base + (size_t)km * 4096 + c0 + koct * 8);
            uint4 st;
            st.x = pack_bf2(__expf((float)kv.h[0] - mxr), __expf((float)kv.h[1] - mxr));
            st.y = pack_bf2(__expf((float)kv.h[2] - mxr), __expf((float)kv.h[3] - mxr));
            st.z = pack_bf2(__expf((float)kv.h[4] - mxr), __expf((float)kv.h[5] - mxr));
            st.w = pack_bf2(__expf((float)kv.h[6] - mxr), __expf((float)kv.h[7] - mxr));
            *(uint4*)kdst = st;
        }
        {   // v -> Vsw (already bf16, pure b128 copy into fragment order)
            *(uint4*)vdst = *(const uint4*)(vb + base + (size_t)vec * 4096 + c0 + voct * 8);
        }
        __syncthreads();
        const v16bf a  = *(const v16bf*)&Ksw[((kc * 2 + mt) * 32 + lane) * 16];
        const v16bf bf = *(const v16bf*)&Vsw[((kc * 2 + s) * 32 + lane) * 16];
        acc = __builtin_amdgcn_wmma_f32_16x16x32_bf16(
            false, a, false, bf, (short)0, acc, false, false);
        __syncthreads();
    }
    {
        const int l = lane & 15, hi = lane >> 4;
#pragma unroll
        for (int r = 0; r < 8; ++r)
            Cp[kc * 32 * 36 + (mt * 16 + hi * 8 + r) * 36 + s * 16 + l] = acc[r];
    }
    __syncthreads();
#pragma unroll
    for (int i = 0; i < 4; ++i) {
        const int lin = tid + i * 256;
        const int d = lin >> 5, e = lin & 31;
        const float sv = Cp[d * 36 + e] + Cp[32 * 36 + d * 36 + e];
        ctx[(size_t)(bfi * 8 + h) * 1024 + d * 32 + e] = sv / ksum[srow + d];
    }
}

// ---------------------------------------------------------------------------
// Kernel 4: attn = ctx^T @ q' (per head), out = w_out @ attn + b_out.
// grid (128 ntiles, 32 bf), block 256.  Tsw holds ctx^T (phase A) then the
// per-chunk w_out tile (phase B).  Step-1 results are scattered directly into
// the step-2 B-operand fragment layout (one b128 store per lane per tile).
// ---------------------------------------------------------------------------
__global__ __launch_bounds__(256) void k_out(const __bf16* __restrict__ qb,
                                             const float* __restrict__ ctx,
                                             const float* __restrict__ wout,
                                             const float* __restrict__ bout,
                                             float* __restrict__ out) {
    __shared__ __align__(32) __bf16 Qsw[8 * 2 * 32 * 16];  // [h][s][lane][e] 16KB
    __shared__ __align__(32) __bf16 Atsw[8 * 2 * 32 * 16]; // [kc][s][lane][e] 16KB
    __shared__ __align__(32) __bf16 Tsw[16 * 32 * 16];     // ctx^T / w tile  16KB

    const int tid   = threadIdx.x;
    const int ntile = blockIdx.x;     // 0..127
    const int bfi   = blockIdx.y;     // 0..31
    const int n0    = ntile * 32;
    const int b     = bfi >> 4, f = bfi & 15;
    const int wave  = tid >> 5, lane = tid & 31;
    const int l     = lane & 15, hi = lane >> 4;

    // ---- load q' into B-operand fragment order ----
    {
        const int qh = tid >> 5, qc = tid & 31;           // head, column
        const int qs = qc >> 4, ql = qc & 15;
        const __bf16* qcol = qb + (size_t)bfi * (256 * 4096)
                           + (size_t)qh * 32 * 4096 + n0 + qc;
        __bf16 qv[32];
#pragma unroll
        for (int d = 0; d < 32; ++d) qv[d] = qcol[(size_t)d * 4096];
        uint4 s0, s1;
        s0.x = pack2h(qv[0], qv[1]);   s0.y = pack2h(qv[2], qv[3]);
        s0.z = pack2h(qv[4], qv[5]);   s0.w = pack2h(qv[6], qv[7]);
        s1.x = pack2h(qv[8], qv[9]);   s1.y = pack2h(qv[10], qv[11]);
        s1.z = pack2h(qv[12], qv[13]); s1.w = pack2h(qv[14], qv[15]);
        __bf16* d0 = &Qsw[(((qh * 2 + qs) * 32) + ql) * 16];
        ((uint4*)d0)[0] = s0; ((uint4*)d0)[1] = s1;
        s0.x = pack2h(qv[16], qv[17]); s0.y = pack2h(qv[18], qv[19]);
        s0.z = pack2h(qv[20], qv[21]); s0.w = pack2h(qv[22], qv[23]);
        s1.x = pack2h(qv[24], qv[25]); s1.y = pack2h(qv[26], qv[27]);
        s1.z = pack2h(qv[28], qv[29]); s1.w = pack2h(qv[30], qv[31]);
        __bf16* d1 = &Qsw[(((qh * 2 + qs) * 32) + 16 + ql) * 16];
        ((uint4*)d1)[0] = s0; ((uint4*)d1)[1] = s1;
    }
    // ---- load ctx^T into A-operand fragment order ----
    {
        const int ch = tid >> 5, cm = tid & 31;           // head, output row (e)
        const int cmt = cm >> 4, clr = cm & 15;
        const float* cc = ctx + (size_t)bfi * 8192 + (size_t)ch * 1024 + cm;
        float cv[32];
#pragma unroll
        for (int d = 0; d < 32; ++d) cv[d] = cc[d * 32];
        uint4 s0, s1;   // row lr: k=d=hi*16+e
        s0.x = pack_bf2(cv[0], cv[1]);   s0.y = pack_bf2(cv[2], cv[3]);
        s0.z = pack_bf2(cv[4], cv[5]);   s0.w = pack_bf2(cv[6], cv[7]);
        s1.x = pack_bf2(cv[8], cv[9]);   s1.y = pack_bf2(cv[10], cv[11]);
        s1.z = pack_bf2(cv[12], cv[13]); s1.w = pack_bf2(cv[14], cv[15]);
        __bf16* d0 = &Tsw[(((ch * 2 + cmt) * 32) + clr) * 16];
        ((uint4*)d0)[0] = s0; ((uint4*)d0)[1] = s1;
        s0.x = pack_bf2(cv[16], cv[17]); s0.y = pack_bf2(cv[18], cv[19]);
        s0.z = pack_bf2(cv[20], cv[21]); s0.w = pack_bf2(cv[22], cv[23]);
        s1.x = pack_bf2(cv[24], cv[25]); s1.y = pack_bf2(cv[26], cv[27]);
        s1.z = pack_bf2(cv[28], cv[29]); s1.w = pack_bf2(cv[30], cv[31]);
        __bf16* d1 = &Tsw[(((ch * 2 + cmt) * 32) + 16 + clr) * 16];
        ((uint4*)d1)[0] = s0; ((uint4*)d1)[1] = s1;
    }
    __syncthreads();

    // ---- phase A: attn[h*32+e][col] = sum_d ctx^T[e][d] * q'[h*32+d][col] ----
#pragma unroll
    for (int i = 0; i < 4; ++i) {
        const int tt = wave * 4 + i;                      // 0..31
        const int h = tt >> 2, sub = tt & 3;
        const int amt = sub & 1, as = sub >> 1;
        const v16bf a  = *(const v16bf*)&Tsw[((h * 2 + amt) * 32 + lane) * 16];
        const v16bf bf = *(const v16bf*)&Qsw[((h * 2 + as) * 32 + lane) * 16];
        v8f acc = v8f{};
        acc = __builtin_amdgcn_wmma_f32_16x16x32_bf16(
            false, a, false, bf, (short)0, acc, false, false);
        // scatter D into step-2 B fragment order: one b128 per lane per tile
        uint4 st;
        st.x = pack_bf2(acc[0], acc[1]); st.y = pack_bf2(acc[2], acc[3]);
        st.z = pack_bf2(acc[4], acc[5]); st.w = pack_bf2(acc[6], acc[7]);
        *(uint4*)&Atsw[((h * 2 + as) * 32 + amt * 16 + l) * 16 + hi * 8] = st;
    }
    __syncthreads();

    // ---- phase B: out = w_out @ attn (K=256, chunks of 32) ----
    v8f acc2[4];
#pragma unroll
    for (int i = 0; i < 4; ++i) acc2[i] = v8f{};

    // w staging task: (row, float4-of-chunk)
    const int wq = tid & 7;
    const int whi = wq & 1, whalf = (wq >> 1) & 1, wk4 = wq >> 2;
    const int wkoff = whi * 16 + whalf * 8 + wk4 * 4;
    const int weoff = whalf * 8 + wk4 * 4;

    for (int kc = 0; kc < 8; ++kc) {
        __syncthreads();                                  // Tsw reuse fence
#pragma unroll
        for (int i = 0; i < 8; ++i) {
            const int row = (tid >> 3) + i * 32;          // 0..255
            const float4 v = *(const float4*)(wout + (size_t)row * 256 + kc * 32 + wkoff);
            uint2 st; st.x = pack_bf2(v.x, v.y); st.y = pack_bf2(v.z, v.w);
            *(uint2*)&Tsw[(((row >> 4) * 32) + (row & 15) + whi * 16) * 16 + weoff] = st;
        }
        __syncthreads();
#pragma unroll
        for (int i = 0; i < 4; ++i) {
            const int tt = wave * 4 + i;                  // m tile 0..15, s 0..1
            const int mtl = tt >> 1, s = tt & 1;
            const v16bf a  = *(const v16bf*)&Tsw[(mtl * 32 + lane) * 16];
            const v16bf bf = *(const v16bf*)&Atsw[((kc * 2 + s) * 32 + lane) * 16];
            acc2[i] = __builtin_amdgcn_wmma_f32_16x16x32_bf16(
                false, a, false, bf, (short)0, acc2[i], false, false);
        }
    }

    // ---- epilogue: + bias, write to (b,c,f,h,w) ----
    float* ob = out + (size_t)b * 16777216 + (size_t)f * 4096;  // ob[c*65536+n]
#pragma unroll
    for (int i = 0; i < 4; ++i) {
        const int tt = wave * 4 + i;
        const int m_base = (tt >> 1) * 16, nco = (tt & 1) * 16;
#pragma unroll
        for (int r = 0; r < 8; ++r) {
            const int c = m_base + hi * 8 + r;
            ob[(size_t)c * 65536 + n0 + nco + l] = acc2[i][r] + bout[c];
        }
    }
}

// ---------------------------------------------------------------------------
extern "C" void kernel_launch(void* const* d_in, const int* in_sizes, int n_in,
                              void* d_out, int out_size, void* d_ws, size_t ws_size,
                              hipStream_t stream) {
    const float* x    = (const float*)d_in[0];
    const float* wqkv = (const float*)d_in[1];
    const float* wout = (const float*)d_in[2];
    const float* bout = (const float*)d_in[3];
    float* out = (float*)d_out;

    // workspace layout (bytes): qb 64MiB | kb 64MiB | vb 64MiB | kmax | ksum | ctx
    char* ws = (char*)d_ws;
    __bf16* qb  = (__bf16*)(ws);
    __bf16* kb  = (__bf16*)(ws + (size_t)67108864);
    __bf16* vb  = (__bf16*)(ws + (size_t)134217728);
    float* kmax = (float*)(ws + (size_t)201326592);
    float* ksum = kmax + 8192;
    float* ctx  = ksum + 8192;

    k_qkv<<<dim3(16, 24, 32), 256, 0, stream>>>(x, wqkv, qb, kb, vb);
    k_stats<<<dim3(8192), 256, 0, stream>>>(kb, kmax, ksum);
    k_ctx<<<dim3(256), 256, 0, stream>>>(kb, vb, kmax, ksum, ctx);
    k_out<<<dim3(128, 32), 256, 0, stream>>>(qb, ctx, wout, bout, out);
}
